// SparseConv1D_34119220199567
// MI455X (gfx1250) — matrix operand order repaired and checked by reading, not compiled, not run on hardware
//
#include <hip/hip_runtime.h>
#include <stdint.h>

typedef __attribute__((ext_vector_type(8)))  _Float16 v8h;
typedef __attribute__((ext_vector_type(16))) _Float16 v16h;
typedef __attribute__((ext_vector_type(8)))  float    v8f;
typedef __attribute__((__vector_size__(4 * sizeof(int)))) int v4i;

#define L_IN    4096
#define PAD_LO  512
#define PAD_HI  256
#define L_PAD   (L_IN + PAD_LO + PAD_HI)   /* 4864 halfs per padded row */
#define NCH     64
#define NTAPS   32
#define NB      16
#define LTILE   128                        /* output columns per block */
#define TILE_U32 (NCH * (LTILE / 2))       /* 4096 u32 = 16 KB per buffer */

#if defined(__AMDGCN__) && __has_builtin(__builtin_amdgcn_global_load_async_to_lds_b32)
#define HAVE_ASYNC_LDS 1
typedef __attribute__((address_space(1))) int* glob_i32_ptr;
typedef __attribute__((address_space(3))) int* lds_i32_ptr;
#if __has_builtin(__builtin_amdgcn_global_load_async_to_lds_b128)
#define HAVE_ASYNC_LDS_B128 1
typedef __attribute__((address_space(1))) v4i* glob_v4i_ptr;
typedef __attribute__((address_space(3))) v4i* lds_v4i_ptr;
#endif
#endif

__constant__ int SK_IND_D[NTAPS] = {
    -512,-256,-128,-96,-64,-48,-32,-24,-16,-12,-8,-6,-4,-3,-2,-1,
       0,   1,   2,  3,  4,  6,  8, 12, 16, 24, 32, 48, 64, 96,128,256};

// ---------------------------------------------------------------------------
// Kernel 1: fp32 -> f16 convert of x into a zero-padded layout:
//   xp[b][i][p], p in [0, L_PAD), value = x[b][i][p-512] or 0.
// ---------------------------------------------------------------------------
__global__ void cvt_x_kernel(const float* __restrict__ x, uint32_t* __restrict__ xp32) {
    size_t u = (size_t)blockIdx.x * blockDim.x + threadIdx.x;
    const size_t totalU = (size_t)NB * NCH * (L_PAD / 2);
    if (u >= totalU) return;
    size_t row = u / (L_PAD / 2);            // b*64 + i
    int    pu  = (int)(u % (L_PAD / 2));
    int    p0  = pu * 2;
    int    l0  = p0 - PAD_LO;
    const float* xr = x + row * (size_t)L_IN;
    float f0 = (l0     >= 0 && l0     < L_IN) ? xr[l0]     : 0.0f;
    float f1 = (l0 + 1 >= 0 && l0 + 1 < L_IN) ? xr[l0 + 1] : 0.0f;
    union { uint32_t u32; _Float16 h[2]; } pk;
    pk.h[0] = (_Float16)f0;
    pk.h[1] = (_Float16)f1;
    xp32[u] = pk.u32;
}

// ---------------------------------------------------------------------------
// Kernel 2: weight transpose + convert: wt[k][o][i] = (f16) w[o][i][k]
// ---------------------------------------------------------------------------
__global__ void cvt_w_kernel(const float* __restrict__ w, _Float16* __restrict__ wt) {
    int idx = blockIdx.x * blockDim.x + threadIdx.x;   // flat [k][o][i]
    if (idx >= NTAPS * NCH * NCH) return;
    int i = idx & 63;
    int o = (idx >> 6) & 63;
    int k = idx >> 12;
    wt[idx] = (_Float16)w[((size_t)o * NCH + i) * NTAPS + k];
}

// ---------------------------------------------------------------------------
// Stage one 64 x 128-half B tile into LDS.
//  - 16B-aligned tap: 4x async b128 DMA global->LDS      (ASYNCcnt)
//  - even tap       : 16x async b32 DMA global->LDS      (ASYNCcnt)
//  - odd tap        : manual funnel-shifted copy (DMA cannot re-align halfs)
// All branches are wave-uniform (tap offset is uniform).
// ---------------------------------------------------------------------------
__device__ __forceinline__ void stage_tile(uint32_t* __restrict__ bt,
                                           const _Float16* __restrict__ xb,
                                           int srow, int scol0, int s0) {
    const uint32_t* srcR = (const uint32_t*)(xb + (size_t)srow * L_PAD);
    const uint32_t* src  = srcR + (s0 >> 1) + scol0;
    uint32_t*       dst  = bt + srow * (LTILE / 2) + scol0;
    if ((s0 & 1) == 0) {
#if defined(HAVE_ASYNC_LDS)
#if defined(HAVE_ASYNC_LDS_B128)
        if ((s0 & 7) == 0) {   // source 16B-aligned: wide DMA
            #pragma unroll
            for (int t = 0; t < 4; ++t) {
                __builtin_amdgcn_global_load_async_to_lds_b128(
                    (glob_v4i_ptr)(src + 4 * t),
                    (lds_v4i_ptr)(dst + 4 * t),
                    0, 0);
            }
        } else
#endif
        {
            #pragma unroll
            for (int t = 0; t < 16; ++t) {
                __builtin_amdgcn_global_load_async_to_lds_b32(
                    (glob_i32_ptr)(src + t),
                    (lds_i32_ptr)(dst + t),
                    0, 0);
            }
        }
#else
        #pragma unroll
        for (int t = 0; t < 16; ++t) dst[t] = src[t];
#endif
    } else {
        #pragma unroll
        for (int t = 0; t < 16; ++t) {
            uint32_t w0 = src[t];
            uint32_t w1 = src[t + 1];
            dst[t] = (w0 >> 16) | (w1 << 16);
        }
    }
}

__device__ __forceinline__ void wait_async_lds() {
#if defined(HAVE_ASYNC_LDS)
#if __has_builtin(__builtin_amdgcn_s_wait_asynccnt)
    __builtin_amdgcn_s_wait_asynccnt(0);
#else
    asm volatile("s_wait_asynccnt 0x0" ::: "memory");
#endif
#endif
}

// ---------------------------------------------------------------------------
// Kernel 3: WMMA GEMM.  Per block: one batch b, one 128-wide L tile, all 64 O.
// 8 waves; wave wv -> M strip (wv&3)*16, N half (wv>>2)*64 (4 n-subtiles).
// K loop: 32 taps x 2 i-chunks of 32 (f16 WMMA K=32).
// Double-buffered LDS: async-stage tap k+1 while WMMAs chew on tap k.
// ---------------------------------------------------------------------------
__global__ __launch_bounds__(256)
void skconv_wmma_kernel(const _Float16* __restrict__ wt,
                        const _Float16* __restrict__ xp,
                        float* __restrict__ out) {
    __shared__ __attribute__((aligned(32))) uint32_t bt32[2 * TILE_U32]; // 32 KB

    const int tid  = threadIdx.x;
    const int lane = tid & 31;
    const int wv   = tid >> 5;                 // 0..7
    const int bb   = blockIdx.x >> 5;          // batch
    const int lt   = blockIdx.x & 31;          // L tile
    const int l0   = lt * LTILE;

    const int mBase  = (wv & 3) * 16;
    const int nGroup = wv >> 2;                // 0 or 1 (64-col halves)
    const int g      = lane >> 4;              // lane half (A frag K-octet select)
    const int mrow   = lane & 15;

    v8f acc0 = {}, acc1 = {}, acc2 = {}, acc3 = {};

    // staging assignment: 4 threads per i-row, 16 u32 each (128 halfs/row)
    const int srow  = tid >> 2;                // 0..63 (i row)
    const int scol0 = (tid & 3) * 16;          // u32 column base

    const _Float16* xb = xp + ((size_t)bb * NCH) * (size_t)L_PAD;

    // prologue: stage tap 0 into buffer 0
    stage_tile(bt32, xb, srow, scol0, PAD_LO + l0 + SK_IND_D[0]);

    for (int k = 0; k < NTAPS; ++k) {
        wait_async_lds();
        __syncthreads();                       // tile k ready; compute(k-1) done

        // kick off DMA of tap k+1 into the other buffer while we compute
        if (k + 1 < NTAPS) {
            uint32_t* nbuf = bt32 + ((k & 1) ? 0 : TILE_U32);
            stage_tile(nbuf, xb, srow, scol0, PAD_LO + l0 + SK_IND_D[k + 1]);
            // warm L0/L2 for next tap's A strip (global_prefetch_b8)
            __builtin_prefetch(wt + ((size_t)(k + 1) << 12)
                                  + (size_t)(mBase + mrow) * NCH, 0, 1);
        }

        const _Float16* btH = (const _Float16*)(bt32 + ((k & 1) ? TILE_U32 : 0));
        #pragma unroll
        for (int ic = 0; ic < 2; ++ic) {
            const int iChunk = ic * 32;
            // A fragment: wt[k][mBase+mrow][iChunk + (two K-octets by lane half)]
            const _Float16* ap = wt + ((size_t)k << 12)
                                    + (size_t)(mBase + mrow) * NCH + iChunk;
            v8h alo = *(const v8h*)(ap + 8 * g);        // K = 8g .. 8g+7
            v8h ahi = *(const v8h*)(ap + 16 + 8 * g);   // K = 16+8g .. 16+8g+7
            v16h a  = __builtin_shufflevector(alo, ahi,
                        0,1,2,3,4,5,6,7,8,9,10,11,12,13,14,15);
            // B fragments: column N=lane&15, K in same octet order as A (by lane half)
            const _Float16* bp = btH + (size_t)iChunk * LTILE + nGroup * 64 + (lane & 15);
            v16h b0, b1, b2, b3;
            #pragma unroll
            for (int s = 0; s < 16; ++s) {
                const int kk = (s < 8) ? (8 * g + s) : (16 + 8 * g + (s - 8));
                b0[s] = bp[kk * LTILE + 0];
                b1[s] = bp[kk * LTILE + 16];
                b2[s] = bp[kk * LTILE + 32];
                b3[s] = bp[kk * LTILE + 48];
            }
            acc0 = __builtin_amdgcn_wmma_f32_16x16x32_f16(false, a, false, b0,
                                                          (short)0, acc0, false, false);
            acc1 = __builtin_amdgcn_wmma_f32_16x16x32_f16(false, a, false, b1,
                                                          (short)0, acc1, false, false);
            acc2 = __builtin_amdgcn_wmma_f32_16x16x32_f16(false, a, false, b2,
                                                          (short)0, acc2, false, false);
            acc3 = __builtin_amdgcn_wmma_f32_16x16x32_f16(false, a, false, b3,
                                                          (short)0, acc3, false, false);
        }
    }

    // ---- write back D (ISA C/D layout: VGPR r -> M=mBase+8*g+r, N=lane&15) ----
    const int mOut  = mBase + g * 8;
    const int nBase = l0 + nGroup * 64 + (lane & 15);
    #pragma unroll
    for (int r = 0; r < 8; ++r) {
        float* orow = out + ((size_t)bb * NCH + mOut + r) * (size_t)L_IN + nBase;
        orow[0]  = acc0[r];
        orow[16] = acc1[r];
        orow[32] = acc2[r];
        orow[48] = acc3[r];
    }
}

// ---------------------------------------------------------------------------
extern "C" void kernel_launch(void* const* d_in, const int* in_sizes, int n_in,
                              void* d_out, int out_size, void* d_ws, size_t ws_size,
                              hipStream_t stream) {
    const float* x = (const float*)d_in[0];
    const float* w = (const float*)d_in[1];
    float*       out = (float*)d_out;

    // ws layout: [0, 256KB) = wt (f16 [k][o][i]);  then padded x16 (~9.96 MB)
    char* ws  = (char*)d_ws;
    _Float16* wt = (_Float16*)ws;
    const size_t wtBytes = (size_t)NTAPS * NCH * NCH * sizeof(_Float16); // 262144
    _Float16* xp = (_Float16*)(ws + wtBytes);

    {
        int total = NTAPS * NCH * NCH;
        cvt_w_kernel<<<(total + 255) / 256, 256, 0, stream>>>(w, wt);
    }
    {
        size_t totalU = (size_t)NB * NCH * (L_PAD / 2);
        cvt_x_kernel<<<(unsigned)((totalU + 255) / 256), 256, 0, stream>>>(
            x, (uint32_t*)xp);
    }
    skconv_wmma_kernel<<<NB * 32, 256, 0, stream>>>(wt, xp, out);
}
